// SeriesBiLSTM_59339268161909
// MI455X (gfx1250) — matrix operand order repaired and checked by reading, not compiled, not run on hardware
//
#include <hip/hip_runtime.h>
#include <hip/hip_bf16.h>

typedef __bf16 bf16;
typedef __attribute__((ext_vector_type(16))) __bf16 v16bf;
typedef __attribute__((ext_vector_type(8)))  __bf16 v8bf;
typedef __attribute__((ext_vector_type(8)))  float  v8f;

#define HSZ 512
#define SEQ 512
#define TFOR 65
#define PP (64*512)   // one ping-pong h buffer (elements)

extern __shared__ char smem_dyn[];

// ---------------------------------------------------------------------------
// WMMA fragment helpers (gfx1250 wave32; contiguous k order for A/B, C/D per CDNA5 ISA)
// A 16x32 bf16: lane m=l&15, half=l>>4; elems 0..15 = k in [half*16,+16)
//               (two contiguous 16B loads)
// B 32x16 bf16 (stored as W[N][K] row-major = B^T): lane n=l&15, half=l>>4;
//               elems 0..15 = k in [half*16,+16)        (contiguous 32B)
// C/D f32 16x16: reg r -> m = r + 8*half, n = l&15
// ---------------------------------------------------------------------------
static __device__ inline v16bf load_a(const bf16* __restrict__ A, int lda,
                                      int mbase, int kb) {
  int lane = threadIdx.x & 31;
  int m    = mbase + (lane & 15);
  int half = lane >> 4;
  const bf16* p = A + (size_t)m * lda + kb + half * 16;
  v8bf lo = *(const v8bf*)p;
  v8bf hi = *(const v8bf*)(p + 8);
  return __builtin_shufflevector(lo, hi, 0,1,2,3,4,5,6,7,8,9,10,11,12,13,14,15);
}

static __device__ inline v16bf load_b(const bf16* __restrict__ W, int ldw,
                                      int nbase, int kb) {
  int lane = threadIdx.x & 31;
  int n    = nbase + (lane & 15);
  int half = lane >> 4;
  return *(const v16bf*)(W + (size_t)n * ldw + kb + half * 16);
}

static __device__ inline v16bf lds_b(const bf16* __restrict__ brow, int k) {
  return *(const v16bf*)(brow + k);
}

static __device__ inline v8f wmma_bf16(v16bf a, v16bf b, v8f c) {
  return __builtin_amdgcn_wmma_f32_16x16x32_bf16(false, a, false, b,
                                                 (short)0, c, false, false);
}

static __device__ inline void store_c(float* __restrict__ dst, int ld, v8f c) {
  int lane = threadIdx.x & 31;
  int n    = lane & 15;
  int half = lane >> 4;
#pragma unroll
  for (int r = 0; r < 8; ++r) dst[(r + half * 8) * ld + n] = c[r];
}

// ---------------------------------------------------------------------------
// Device-wide barrier (counter memset to 0 on stream before each kernel).
// ---------------------------------------------------------------------------
static __device__ inline void grid_barrier(unsigned* cnt) {
  __syncthreads();
  if (threadIdx.x == 0) {
    __threadfence();
    unsigned nb     = gridDim.x;
    unsigned prev   = atomicAdd(cnt, 1u);
    unsigned target = (prev / nb + 1u) * nb;
    while (__hip_atomic_load(cnt, __ATOMIC_ACQUIRE, __HIP_MEMORY_SCOPE_AGENT) < target)
      __builtin_amdgcn_s_sleep(2);
    __threadfence();
  }
  __syncthreads();
}

static __device__ inline float sigf(float x) { return 1.f / (1.f + __expf(-x)); }

// ---------------------------------------------------------------------------
// One-time copy of this block's 64 gate-rows of W into (padded) LDS.
// LDS row r (0..63) <-> global row (r>>4)*HSZ + j0 + (r&15); 16B chunks.
// Row pad of 8 bf16 (16B) -> row stride 16 mod 256B -> conflict-free b128.
// ---------------------------------------------------------------------------
static __device__ inline void preload_w(const bf16* __restrict__ Wg, int ldw,
                                        int j0, bf16* __restrict__ Wlds, int ldp) {
  int nchunks = ldw >> 3;                    // 16B chunks per row
  for (int u = threadIdx.x; u < 64 * nchunks; u += blockDim.x) {
    int r = u / nchunks, c = u - r * nchunks;
    int grow = (r >> 4) * HSZ + j0 + (r & 15);
    *(uint4*)(Wlds + (size_t)r * ldp + c * 8) =
        *(const uint4*)(Wg + (size_t)grow * ldw + c * 8);
  }
}

// ---------------------------------------------------------------------------
// Gate GEMM, B from LDS-resident weight slice. K unrolled by 2 (K % 64 == 0),
// final iteration peeled so the A prefetch is branch-free in steady state.
// Block owns 16 hidden units across 4 gates; 8 waves = 4(m) x 2(gate-pair).
// ---------------------------------------------------------------------------
static __device__ inline void gates_gemm_lds(const bf16* const* parts,
                                             const int* partK, int nparts,
                                             const bf16* __restrict__ Wlds, int ldp,
                                             float* __restrict__ lds_g) {
  int tid   = threadIdx.x;
  int wave  = tid >> 5;
  int mbase = (wave & 3) * 16;
  int g0    = (wave >> 2) * 2;
  int lane  = tid & 31;
  int half  = lane >> 4;
  const bf16* brow0 = Wlds + (size_t)(g0 * 16 + (lane & 15)) * ldp + half * 16;
  const bf16* brow1 = Wlds + (size_t)((g0 + 1) * 16 + (lane & 15)) * ldp + half * 16;
  v8f acc0 = {0,0,0,0,0,0,0,0};
  v8f acc1 = {0,0,0,0,0,0,0,0};
  int koff = 0;
  for (int pi = 0; pi < nparts; ++pi) {
    const bf16* P = parts[pi];
    const int K = partK[pi];
    v16bf a0 = load_a(P, K, mbase, 0);
    v16bf a1 = load_a(P, K, mbase, 32);
    int kb = 0;
    for (; kb + 64 < K; kb += 64) {
      v16bf an0 = load_a(P, K, mbase, kb + 64);   // unconditional prefetch
      v16bf an1 = load_a(P, K, mbase, kb + 96);
      v16bf b00 = lds_b(brow0, koff + kb);
      v16bf b10 = lds_b(brow1, koff + kb);
      v16bf b01 = lds_b(brow0, koff + kb + 32);
      v16bf b11 = lds_b(brow1, koff + kb + 32);
      acc0 = wmma_bf16(a0, b00, acc0);
      acc1 = wmma_bf16(a0, b10, acc1);
      acc0 = wmma_bf16(a1, b01, acc0);
      acc1 = wmma_bf16(a1, b11, acc1);
      a0 = an0; a1 = an1;
    }
    { // peeled final 64
      v16bf b00 = lds_b(brow0, koff + kb);
      v16bf b10 = lds_b(brow1, koff + kb);
      v16bf b01 = lds_b(brow0, koff + kb + 32);
      v16bf b11 = lds_b(brow1, koff + kb + 32);
      acc0 = wmma_bf16(a0, b00, acc0);
      acc1 = wmma_bf16(a0, b10, acc1);
      acc0 = wmma_bf16(a1, b01, acc0);
      acc1 = wmma_bf16(a1, b11, acc1);
    }
    koff += K;
  }
  store_c(lds_g + ((g0)     * 64 + mbase) * 16, 16, acc0);
  store_c(lds_g + ((g0 + 1) * 64 + mbase) * 16, 16, acc1);
}

// Same structure, B from global (where LDS capacity is spent elsewhere).
static __device__ inline void gates_gemm_glb(const bf16* const* parts,
                                             const int* partK, int nparts,
                                             const bf16* __restrict__ W, int ldw,
                                             int j0, float* __restrict__ lds_g) {
  int tid   = threadIdx.x;
  int wave  = tid >> 5;
  int mbase = (wave & 3) * 16;
  int g0    = (wave >> 2) * 2;
  int n0    = g0 * HSZ + j0;
  int n1    = n0 + HSZ;
  v8f acc0 = {0,0,0,0,0,0,0,0};
  v8f acc1 = {0,0,0,0,0,0,0,0};
  int koff = 0;
  for (int pi = 0; pi < nparts; ++pi) {
    const bf16* P = parts[pi];
    const int K = partK[pi];
    v16bf a0 = load_a(P, K, mbase, 0);
    v16bf a1 = load_a(P, K, mbase, 32);
    int kb = 0;
    for (; kb + 64 < K; kb += 64) {
      v16bf an0 = load_a(P, K, mbase, kb + 64);
      v16bf an1 = load_a(P, K, mbase, kb + 96);
      v16bf b00 = load_b(W, ldw, n0, koff + kb);
      v16bf b10 = load_b(W, ldw, n1, koff + kb);
      v16bf b01 = load_b(W, ldw, n0, koff + kb + 32);
      v16bf b11 = load_b(W, ldw, n1, koff + kb + 32);
      acc0 = wmma_bf16(a0, b00, acc0);
      acc1 = wmma_bf16(a0, b10, acc1);
      acc0 = wmma_bf16(a1, b01, acc0);
      acc1 = wmma_bf16(a1, b11, acc1);
      a0 = an0; a1 = an1;
    }
    {
      v16bf b00 = load_b(W, ldw, n0, koff + kb);
      v16bf b10 = load_b(W, ldw, n1, koff + kb);
      v16bf b01 = load_b(W, ldw, n0, koff + kb + 32);
      v16bf b11 = load_b(W, ldw, n1, koff + kb + 32);
      acc0 = wmma_bf16(a0, b00, acc0);
      acc1 = wmma_bf16(a0, b10, acc1);
      acc0 = wmma_bf16(a1, b01, acc0);
      acc1 = wmma_bf16(a1, b11, acc1);
    }
    koff += K;
  }
  store_c(lds_g + ((g0)     * 64 + mbase) * 16, 16, acc0);
  store_c(lds_g + ((g0 + 1) * 64 + mbase) * 16, 16, acc1);
}

// Single-tile GEMM (q / out projections): K unrolled by 2, peeled tail.
static __device__ inline v8f tile_gemm(const bf16* __restrict__ A, int lda,
                                       const bf16* __restrict__ W, int ldw,
                                       int mb, int nb, int K) {
  v8f acc = {0,0,0,0,0,0,0,0};
  v16bf a0 = load_a(A, lda, mb, 0);
  v16bf a1 = load_a(A, lda, mb, 32);
  int kb = 0;
  for (; kb + 64 < K; kb += 64) {
    v16bf an0 = load_a(A, lda, mb, kb + 64);
    v16bf an1 = load_a(A, lda, mb, kb + 96);
    v16bf b0 = load_b(W, ldw, nb, kb);
    v16bf b1 = load_b(W, ldw, nb, kb + 32);
    acc = wmma_bf16(a0, b0, acc);
    acc = wmma_bf16(a1, b1, acc);
    a0 = an0; a1 = an1;
  }
  {
    v16bf b0 = load_b(W, ldw, nb, kb);
    v16bf b1 = load_b(W, ldw, nb, kb + 32);
    acc = wmma_bf16(a0, b0, acc);
    acc = wmma_bf16(a1, b1, acc);
  }
  return acc;
}

// i,f,g,o -> (h,c); c lives in registers (ownership mapping is step-invariant)
static __device__ inline void lstm_act(const float* __restrict__ lds_g,
                                       const float* __restrict__ bias, int j0,
                                       float* creg, bf16* __restrict__ hout,
                                       bf16* __restrict__ yrow, int yld) {
  int tid = threadIdx.x;
#pragma unroll
  for (int e = 0; e < 4; ++e) {
    int p = e * 256 + tid;            // 0..1023 over (batch=64, jj=16)
    int bm = p >> 4, jj = p & 15, j = j0 + jj;
    float gi = lds_g[(0 * 64 + bm) * 16 + jj] + bias[j];
    float gf = lds_g[(1 * 64 + bm) * 16 + jj] + bias[HSZ + j];
    float gg = lds_g[(2 * 64 + bm) * 16 + jj] + bias[2 * HSZ + j];
    float go = lds_g[(3 * 64 + bm) * 16 + jj] + bias[3 * HSZ + j];
    float c  = sigf(gf) * creg[e] + sigf(gi) * tanhf(gg);
    creg[e]  = c;
    bf16 hb  = (bf16)(sigf(go) * tanhf(c));
    hout[bm * HSZ + j] = hb;
    if (yrow) yrow[bm * yld + j] = hb;
  }
}

// ---------------------------------------------------------------------------
// Prep kernels
// ---------------------------------------------------------------------------
__global__ void pack_w(const float* __restrict__ A, int KA,
                       const float* __restrict__ B, int KB,
                       bf16* __restrict__ dst) {
  int row = blockIdx.y;
  int ld  = KA + KB;
  for (int k = blockIdx.x * blockDim.x + threadIdx.x; k < ld;
       k += gridDim.x * blockDim.x) {
    float v = (k < KA) ? A[(size_t)row * KA + k] : B[(size_t)row * KB + (k - KA)];
    dst[(size_t)row * ld + k] = (bf16)v;
  }
}

__global__ void in_proj(const float* __restrict__ X, const float* __restrict__ w,
                        const float* __restrict__ b, bf16* __restrict__ Xp) {
  size_t idx = (size_t)blockIdx.x * blockDim.x + threadIdx.x; // over 576*64*512
  int h = idx & 511;
  size_t bt = idx >> 9;
  int bat = bt & 63;
  int t   = (int)(bt >> 6);
  if (t < 576) Xp[idx] = (bf16)(X[(size_t)bat * 576 + t] * w[h] + b[h]);
}

// ---------------------------------------------------------------------------
// Persistent bidirectional LSTM layer: 64 blocks = 2 dirs x 32 hidden slices.
// Weight slice LDS-resident for all 512 steps; one grid barrier per step.
// Dynamic LDS: [Wlds 64 x (Kx+512+8) bf16][gates 4*64*16 f32]
// ---------------------------------------------------------------------------
__global__ __launch_bounds__(256) void enc_layer(
    const bf16* __restrict__ Xin, int Kx,
    const bf16* __restrict__ Wfw, const bf16* __restrict__ Wbw,
    const float* __restrict__ bfw, const float* __restrict__ bbw,
    bf16* __restrict__ hbuf,               // [2 dir][2 pp][64*512]
    bf16* __restrict__ yout,               // [S][64][1024]
    unsigned* __restrict__ cnt) {
  const int dir   = blockIdx.x >> 5;
  const int slice = blockIdx.x & 31;
  const int j0    = slice * 16;
  const bf16*  W    = dir ? Wbw : Wfw;
  const float* bias = dir ? bbw : bfw;
  const int ldw = Kx + HSZ;
  const int ldp = ldw + 8;                 // +16B pad -> conflict-free ds_load_b128
  bf16* hdir = hbuf + (size_t)dir * 2 * PP;

  bf16*  Wlds  = (bf16*)smem_dyn;
  float* lds_g = (float*)(smem_dyn + (size_t)64 * ldp * 2);
  const int tid = threadIdx.x;

  preload_w(W, ldw, j0, Wlds, ldp);        // once; replayed 512x from LDS

  float creg[4];
#pragma unroll
  for (int e = 0; e < 4; ++e) {
    creg[e] = 0.f;
    int p = e * 256 + tid;
    int bm = p >> 4, jj = p & 15;
    hdir[bm * HSZ + j0 + jj] = (bf16)0.f;  // pp buffer 0
  }
  grid_barrier(cnt);

  for (int t = 0; t < SEQ; ++t) {
    const int te = dir ? (SEQ - 1 - t) : t;
    // pull next step's input row toward the WGP while this step computes
    if (t + 1 < SEQ) {
      const int tn = dir ? (SEQ - 2 - t) : (t + 1);
      const bf16* xn = Xin + (size_t)tn * 64 * Kx;
      for (int u = tid; u < Kx; u += 256)
        __builtin_prefetch(xn + (size_t)u * 64, 0, 1);
    }
    const bf16* parts[2] = {Xin + (size_t)te * 64 * Kx, hdir + (size_t)(t & 1) * PP};
    int partK[2] = {Kx, HSZ};
    gates_gemm_lds(parts, partK, 2, Wlds, ldp, lds_g);
    __syncthreads();
    lstm_act(lds_g, bias, j0, creg, hdir + (size_t)((t + 1) & 1) * PP,
             yout + (size_t)te * 64 * 1024 + dir * HSZ, 1024);
    grid_barrier(cnt);
  }
}

// ---------------------------------------------------------------------------
// Persistent decoder: 32 blocks, phases separated by grid barriers.
// Cell-0 weight slice (K=1536, the big one) is LDS-resident for all 65 steps.
// Dynamic LDS: [W0lds 64 x 1544 bf16][gates 4*64*16 f32]
// ---------------------------------------------------------------------------
__global__ __launch_bounds__(256) void decoder(
    const bf16* __restrict__ Xp,                      // [576][64][512]
    const bf16* __restrict__ W0, const float* __restrict__ b0,   // [2048][1536]
    const bf16* __restrict__ W1, const float* __restrict__ b1,   // [2048][1024]
    const bf16* __restrict__ Wai, const float* __restrict__ bai, // [1024][512]
    const bf16* __restrict__ Wao, const float* __restrict__ bao, // [512][1536]
    const float* __restrict__ dh0, const float* __restrict__ dc0,// [2][512]
    const bf16* __restrict__ ctxseq,                  // [512][64][1024]
    bf16* __restrict__ h0, bf16* __restrict__ h1, bf16* __restrict__ feed, // [2][64*512]
    bf16* __restrict__ q,                             // [64][1024]
    float* __restrict__ scores,                       // [512][64]
    bf16* __restrict__ ctxbuf,                        // [64][1536]
    float* __restrict__ out,                          // [64][65][512]
    unsigned* __restrict__ cnt) {
  const int j0   = blockIdx.x * 16;
  const int tid  = threadIdx.x;
  const int wave = tid >> 5;
  const int ldp0 = 1536 + 8;

  bf16*  W0lds = (bf16*)smem_dyn;
  float* lds_g = (float*)(smem_dyn + (size_t)64 * ldp0 * 2);
  __shared__ float red[256];

  preload_w(W0, 1536, j0, W0lds, ldp0);

  float c0reg[4], c1reg[4];
#pragma unroll
  for (int e = 0; e < 4; ++e) {
    int p = e * 256 + tid;
    int bm = p >> 4, jj = p & 15, j = j0 + jj;
    c0reg[e] = dc0[j];
    c1reg[e] = dc0[HSZ + j];
    h0[bm * HSZ + j]   = (bf16)dh0[j];
    h1[bm * HSZ + j]   = (bf16)dh0[HSZ + j];
    feed[bm * HSZ + j] = (bf16)0.f;
  }
  grid_barrier(cnt);

  for (int t = 0; t < TFOR; ++t) {
    const int par = t & 1, nxt = par ^ 1;
    const bf16* xt = Xp + (size_t)(SEQ - 1 + t) * 64 * HSZ;

    { // phase 1: decoder cell 0   inp = [x_t | feed], own h0 (LDS weights)
      const bf16* parts[3] = {xt, feed + (size_t)par * PP, h0 + (size_t)par * PP};
      int partK[3] = {HSZ, HSZ, HSZ};
      gates_gemm_lds(parts, partK, 3, W0lds, ldp0, lds_g);
      __syncthreads();
      lstm_act(lds_g, b0, j0, c0reg, h0 + (size_t)nxt * PP, nullptr, 0);
      grid_barrier(cnt);
    }
    { // phase 2: decoder cell 1   inp = h0_new, own h1 (global weights)
      const bf16* parts[2] = {h0 + (size_t)nxt * PP, h1 + (size_t)par * PP};
      int partK[2] = {HSZ, HSZ};
      gates_gemm_glb(parts, partK, 2, W1, 1024, j0, lds_g);
      __syncthreads();
      lstm_act(lds_g, b1, j0, c1reg, h1 + (size_t)nxt * PP, nullptr, 0);
      grid_barrier(cnt);
    }
    { // phase 3: q = h1_new @ Wa_in^T + ba_in  -> [64,1024] bf16 (WMMA)
      int w = blockIdx.x * 8 + wave;          // 0..255, one 16x16 tile each
      int mb = (w & 3) * 16, nb = (w >> 2) * 16;
      v8f acc = tile_gemm(h1 + (size_t)nxt * PP, HSZ, Wai, HSZ, mb, nb, HSZ);
      int lane = tid & 31, nn = nb + (lane & 15), half = lane >> 4;
#pragma unroll
      for (int r = 0; r < 8; ++r) {
        int bmr = mb + r + half * 8;
        q[bmr * 1024 + nn] = (bf16)(acc[r] + bai[nn]);
      }
      grid_barrier(cnt);
    }
    { // phase 4: scores[s][b] = <context[s][b][:], q[b][:]>  (VALU, bf16->f32)
      int s0 = blockIdx.x * 16;
#pragma unroll
      for (int e = 0; e < 4; ++e) {
        int pi = e * 256 + tid;               // 0..1023 = 16 s x 64 b
        int s = s0 + (pi >> 6), bb = pi & 63;
        const bf16* crow = ctxseq + ((size_t)s * 64 + bb) * 1024;
        const bf16* qrow = q + (size_t)bb * 1024;
        float accd = 0.f;
        for (int d = 0; d < 1024; d += 8) {
          v8bf cv = *(const v8bf*)(crow + d);
          v8bf qv = *(const v8bf*)(qrow + d);
#pragma unroll
          for (int u = 0; u < 8; ++u) accd += (float)cv[u] * (float)qv[u];
        }
        scores[s * 64 + bb] = accd;
      }
      grid_barrier(cnt);
    }
    { // phase 5: softmax over s, per batch column (2 columns per block)
      int halfT = tid >> 7, lt = tid & 127;
      int bb = blockIdx.x * 2 + halfT;
      float vals[4], vmax = -1e30f;
#pragma unroll
      for (int e = 0; e < 4; ++e) {
        vals[e] = scores[(lt + 128 * e) * 64 + bb];
        vmax = fmaxf(vmax, vals[e]);
      }
      red[tid] = vmax; __syncthreads();
      for (int off = 64; off > 0; off >>= 1) {
        if (lt < off) red[tid] = fmaxf(red[tid], red[tid + off]);
        __syncthreads();
      }
      vmax = red[halfT * 128]; __syncthreads();
      float sum = 0.f;
#pragma unroll
      for (int e = 0; e < 4; ++e) { vals[e] = __expf(vals[e] - vmax); sum += vals[e]; }
      red[tid] = sum; __syncthreads();
      for (int off = 64; off > 0; off >>= 1) {
        if (lt < off) red[tid] += red[tid + off];
        __syncthreads();
      }
      float inv = 1.f / red[halfT * 128]; __syncthreads();
#pragma unroll
      for (int e = 0; e < 4; ++e) scores[(lt + 128 * e) * 64 + bb] = vals[e] * inv;
      grid_barrier(cnt);
    }
    { // phase 6: ctx[b][d] = sum_s a[s][b]*context[s][b][d]; build [ctx | h1]
      int halfT = tid >> 7, lt = tid & 127;
      int bb = blockIdx.x * 2 + halfT;
      int d0 = lt * 8;
      float a8[8] = {0,0,0,0,0,0,0,0};
      for (int s = 0; s < SEQ; ++s) {
        float as = scores[s * 64 + bb];
        v8bf cv = *(const v8bf*)(ctxseq + ((size_t)s * 64 + bb) * 1024 + d0);
#pragma unroll
        for (int u = 0; u < 8; ++u) a8[u] += as * (float)cv[u];
      }
#pragma unroll
      for (int u = 0; u < 8; ++u) ctxbuf[bb * 1536 + d0 + u] = (bf16)a8[u];
      const bf16* h1n = h1 + (size_t)nxt * PP;
#pragma unroll
      for (int r = 0; r < 4; ++r) {
        int j = lt + 128 * r;
        ctxbuf[bb * 1536 + 1024 + j] = h1n[bb * HSZ + j];
      }
      grid_barrier(cnt);
    }
    { // phase 7: out = tanh([ctx|h1] @ Wa_out^T + ba_out)  (WMMA, 128 tiles)
      int w = blockIdx.x * 8 + wave;
      if (w < 128) {
        int mb = (w & 3) * 16, nb = (w >> 2) * 16;
        v8f acc = tile_gemm(ctxbuf, 1536, Wao, 1536, mb, nb, 1536);
        int lane = tid & 31, nn = nb + (lane & 15), half = lane >> 4;
        bf16* fout = feed + (size_t)nxt * PP;
#pragma unroll
        for (int r = 0; r < 8; ++r) {
          int bmr = mb + r + half * 8;
          float v = tanhf(acc[r] + bao[nn]);
          out[((size_t)bmr * TFOR + t) * HSZ + nn] = v;   // [B][Tf][H]
          fout[bmr * HSZ + nn] = (bf16)v;
        }
      }
      grid_barrier(cnt);
    }
  }
}

// ---------------------------------------------------------------------------
// Host launcher
// ---------------------------------------------------------------------------
extern "C" void kernel_launch(void* const* d_in, const int* in_sizes, int n_in,
                              void* d_out, int out_size, void* d_ws, size_t ws_size,
                              hipStream_t stream) {
  (void)in_sizes; (void)n_in; (void)out_size; (void)ws_size;
  char* ws = (char*)d_ws;
  size_t off = 0;
  auto alloc = [&](size_t bytes) -> char* {
    off = (off + 255) & ~(size_t)255;
    char* p = ws + off;
    off += bytes;
    return p;
  };

  bf16* Xp  = (bf16*)alloc(576ull * 64 * 512 * 2);
  bf16* W0f = (bf16*)alloc(2048ull * 1024 * 2);
  bf16* W0b = (bf16*)alloc(2048ull * 1024 * 2);
  bf16* W1f = (bf16*)alloc(2048ull * 1536 * 2);
  bf16* W1b = (bf16*)alloc(2048ull * 1536 * 2);
  bf16* WD0 = (bf16*)alloc(2048ull * 1536 * 2);
  bf16* WD1 = (bf16*)alloc(2048ull * 1024 * 2);
  bf16* WAI = (bf16*)alloc(1024ull * 512 * 2);
  bf16* WAO = (bf16*)alloc(512ull * 1536 * 2);
  bf16* Y0  = (bf16*)alloc(512ull * 64 * 1024 * 2);
  bf16* CTX = (bf16*)alloc(512ull * 64 * 1024 * 2);
  bf16* HE  = (bf16*)alloc(2ull * 2 * PP * 2);
  bf16* H0  = (bf16*)alloc(2ull * PP * 2);
  bf16* H1  = (bf16*)alloc(2ull * PP * 2);
  bf16* FEED= (bf16*)alloc(2ull * PP * 2);
  bf16* Q   = (bf16*)alloc(64ull * 1024 * 2);
  float* SC = (float*)alloc(512ull * 64 * 4);
  bf16* CB  = (bf16*)alloc(64ull * 1536 * 2);
  unsigned* CNT = (unsigned*)alloc(256);

  // weight packing: [Wih | Whh] -> bf16, native WMMA-B layout ([N][K] row-major)
  pack_w<<<dim3(2, 2048), 256, 0, stream>>>((const float*)d_in[3],  512, (const float*)d_in[4],  512, W0f);
  pack_w<<<dim3(2, 2048), 256, 0, stream>>>((const float*)d_in[6],  512, (const float*)d_in[7],  512, W0b);
  pack_w<<<dim3(2, 2048), 256, 0, stream>>>((const float*)d_in[9],  1024,(const float*)d_in[10], 512, W1f);
  pack_w<<<dim3(2, 2048), 256, 0, stream>>>((const float*)d_in[12], 1024,(const float*)d_in[13], 512, W1b);
  pack_w<<<dim3(2, 2048), 256, 0, stream>>>((const float*)d_in[15], 1024,(const float*)d_in[16], 512, WD0);
  pack_w<<<dim3(2, 2048), 256, 0, stream>>>((const float*)d_in[18], 512, (const float*)d_in[19], 512, WD1);
  pack_w<<<dim3(2, 1024), 256, 0, stream>>>((const float*)d_in[23], 512, nullptr, 0, WAI);
  pack_w<<<dim3(2, 512),  256, 0, stream>>>((const float*)d_in[25], 1536,nullptr, 0, WAO);

  // in-projection -> Xp (time-major bf16)
  in_proj<<<(576 * 64 * 512) / 256, 256, 0, stream>>>(
      (const float*)d_in[0], (const float*)d_in[1], (const float*)d_in[2], Xp);

  // dynamic LDS sizes: weight slice (padded) + gate tile buffer
  const size_t shm_l0  = (size_t)64 * (1024 + 512 + 8) * 2 + 4 * 64 * 16 * 4; // 214016
  const size_t shm_l1  = (size_t)64 * (1536 + 512 + 8) * 2 + 4 * 64 * 16 * 4; // 279552
  const size_t shm_dec = (size_t)64 * (1536 + 8) * 2 + 4 * 64 * 16 * 4;       // 214016

  // encoder layer 0 (bidirectional, persistent, grid-synced)
  hipMemsetAsync(CNT, 0, 256, stream);
  enc_layer<<<64, 256, shm_l0, stream>>>(Xp, 512, W0f, W0b,
                                         (const float*)d_in[5], (const float*)d_in[8],
                                         HE, Y0, CNT);
  // encoder layer 1
  hipMemsetAsync(CNT, 0, 256, stream);
  enc_layer<<<64, 256, shm_l1, stream>>>(Y0, 1024, W1f, W1b,
                                         (const float*)d_in[11], (const float*)d_in[14],
                                         HE, CTX, CNT);
  // decoder + attention (persistent)
  hipMemsetAsync(CNT, 0, 256, stream);
  decoder<<<32, 256, shm_dec, stream>>>(Xp,
                                        WD0, (const float*)d_in[17],
                                        WD1, (const float*)d_in[20],
                                        WAI, (const float*)d_in[24],
                                        WAO, (const float*)d_in[26],
                                        (const float*)d_in[21], (const float*)d_in[22],
                                        CTX, H0, H1, FEED, Q, SC, CB,
                                        (float*)d_out, CNT);
}
